// Block_mamba_62646392979995
// MI455X (gfx1250) — matrix operand order repaired and checked by reading, not compiled, not run on hardware
//
#include <hip/hip_runtime.h>
#include <hip/hip_bf16.h>
#include <math.h>

#define DIMC     256
#define DSTATE   64
#define DINNER   512
#define DTRANK   16
#define BATCH    2
#define NTOK     2048
#define BLROWS   (BATCH * NTOK)          // 4096
#define XDBLC    (DTRANK + 2 * DSTATE)   // 144 (logical)
#define XDBLP    192                     // padded to multiple of 64 for WMMA tiles
#define LAMBDA_S 0.01f

typedef __attribute__((ext_vector_type(16))) _Float16     v16h;
typedef __attribute__((ext_vector_type(8)))  float        v8f;
typedef __attribute__((ext_vector_type(4)))  unsigned int u32x4;
typedef __attribute__((ext_vector_type(8)))  unsigned int u32x8;

// ---------------------------------------------------------------------------
// Weight transpose + fp32 -> f16 with column padding:
//   W[O][C] -> Wt[C][Opad], Wt[c][o] = (o < O) ? W[o][c] : 0
// ---------------------------------------------------------------------------
__global__ void k_transpose_f16(const float* __restrict__ W, _Float16* __restrict__ Wt,
                                int O, int C, int Opad) {
    int i = blockIdx.x * 256 + threadIdx.x;
    if (i >= C * Opad) return;
    int c = i / Opad, o = i % Opad;
    Wt[i] = (o < O) ? (_Float16)W[(size_t)o * C + c] : (_Float16)0.f;
}

// ---------------------------------------------------------------------------
// LayerNorm over 256 channels; one block per row; optional f16 / f32 outputs
// ---------------------------------------------------------------------------
__global__ void k_layernorm(const float* __restrict__ x, const float* __restrict__ w,
                            const float* __restrict__ b,
                            _Float16* __restrict__ outH, float* __restrict__ outF) {
    int row = blockIdx.x, t = threadIdx.x;
    __shared__ float red[DIMC];
    float v = x[(size_t)row * DIMC + t];
    red[t] = v; __syncthreads();
    for (int s = 128; s > 0; s >>= 1) { if (t < s) red[t] += red[t + s]; __syncthreads(); }
    float mean = red[0] / (float)DIMC; __syncthreads();
    float dv = v - mean;
    red[t] = dv * dv; __syncthreads();
    for (int s = 128; s > 0; s >>= 1) { if (t < s) red[t] += red[t + s]; __syncthreads(); }
    float var = red[0] / (float)DIMC;
    float r = dv * rsqrtf(var + 1e-5f) * w[t] + b[t];
    size_t idx = (size_t)row * DIMC + t;
    if (outH) outH[idx] = (_Float16)r;
    if (outF) outF[idx] = r;
}

// ---------------------------------------------------------------------------
// TDM: stage a 32 x 64 f16 tile (rows k0..k0+31, cols nblk..nblk+63) of a
// row-major [K x N] f16 matrix into LDS.  Descriptor per CDNA5 ISA ch.8.
// ---------------------------------------------------------------------------
__device__ __forceinline__ void tdm_load_tile(const _Float16* gsrc, unsigned lds_byte_addr,
                                              int cols_remaining, int rowstride_elems) {
    unsigned long long ga = (unsigned long long)(uintptr_t)gsrc;
    unsigned td0 = (unsigned)cols_remaining;
    u32x4 g0;
    g0[0] = 1u;                                            // count=1, user mode
    g0[1] = lds_byte_addr;                                 // lds_addr
    g0[2] = (unsigned)ga;                                  // global_addr lo
    g0[3] = (unsigned)((ga >> 32) & 0x01FFFFFFull) | 0x80000000u;  // ga hi | type=2
    u32x8 g1;
    g1[0] = 0x00010000u;                                   // data_size = 2 bytes
    g1[1] = (td0 & 0xFFFFu) << 16;                         // tensor_dim0 lo16
    g1[2] = ((td0 >> 16) & 0xFFFFu) | (32u << 16);         // tensor_dim0 hi | tensor_dim1=32
    g1[3] = (64u << 16);                                   // tile_dim0 = 64
    g1[4] = 32u;                                           // tile_dim1 = 32 (tile_dim2=0)
    g1[5] = (unsigned)rowstride_elems;                     // tensor_dim0_stride lo32
    g1[6] = 0u;
    g1[7] = 0u;
    asm volatile("tensor_load_to_lds %0, %1" :: "s"(g0), "s"(g1) : "memory");
}

// ---------------------------------------------------------------------------
// WMMA f16 GEMM:  C[M,N] = A[M,K]*B[K,N] (+residual), A/B f16 row-major.
// Block = 8 waves stacked over M (128 rows); each wave computes 16x64 via
// 4 back-to-back WMMAs sharing one A fragment.  B tile staged in LDS by TDM,
// double buffered, synchronized with s_wait_tensorcnt.
// Requires M % 128 == 0, N % 64 == 0, K % 32 == 0.
// ---------------------------------------------------------------------------
__global__ void k_gemm_wmma(const _Float16* __restrict__ A, const _Float16* __restrict__ B,
                            float* __restrict__ C, const float* __restrict__ residual,
                            int M, int N, int K) {
    __shared__ __align__(32) _Float16 Bt[2][32 * 64];
    int lane = threadIdx.x & 31;
    int wv   = threadIdx.x >> 5;
    int m0   = blockIdx.x * 128 + wv * 16;
    int nblk = blockIdx.y * 64;
    int half = lane >> 4;

    const _Float16* Arow = A + (size_t)(m0 + (lane & 15)) * K;

    if (wv == 0)
        tdm_load_tile(B + nblk, (unsigned)(uintptr_t)&Bt[0][0], N - nblk, N);

    v8f acc0 = {}, acc1 = {}, acc2 = {}, acc3 = {};
    for (int k0 = 0; k0 < K; k0 += 32) {
        int cur = (k0 >> 5) & 1;
        if (wv == 0) __builtin_amdgcn_s_wait_tensorcnt(0);
        __syncthreads();                     // tile[cur] visible; tile[cur^1] free
        if (wv == 0 && k0 + 32 < K)
            tdm_load_tile(B + (size_t)(k0 + 32) * N + nblk,
                          (unsigned)(uintptr_t)&Bt[cur ^ 1][0], N - nblk, N);

        v16h af;
#pragma unroll
        for (int j = 0; j < 8; ++j) {
            // A 16x32 f16 frag: VGPR j holds K = (j/4)*16 + half*8 + (j%4)*2 (+1)
            int kl = ((j >> 2) << 4) + (half << 3) + ((j & 3) << 1);
            af[2 * j]     = Arow[k0 + kl];
            af[2 * j + 1] = Arow[k0 + kl + 1];
        }
        const _Float16* brow = &Bt[cur][lane & 15];   // B frag: lane%16 = N, elem s -> K (same order as A)
        v16h b0, b1, b2, b3;
#pragma unroll
        for (int s = 0; s < 16; ++s) {
            int kb = (s < 8) ? (half << 3) + s : 16 + (half << 3) + (s - 8);
            b0[s] = brow[kb * 64];
            b1[s] = brow[kb * 64 + 16];
            b2[s] = brow[kb * 64 + 32];
            b3[s] = brow[kb * 64 + 48];
        }
        acc0 = __builtin_amdgcn_wmma_f32_16x16x32_f16(false, af, false, b0, (short)0, acc0, false, false);
        acc1 = __builtin_amdgcn_wmma_f32_16x16x32_f16(false, af, false, b1, (short)0, acc1, false, false);
        acc2 = __builtin_amdgcn_wmma_f32_16x16x32_f16(false, af, false, b2, (short)0, acc2, false, false);
        acc3 = __builtin_amdgcn_wmma_f32_16x16x32_f16(false, af, false, b3, (short)0, acc3, false, false);
    }

    int col = lane & 15;
    float* Cbase = C + (size_t)(m0 + half * 8) * N + nblk + col;
    const float* Rbase = residual ? residual + (size_t)(m0 + half * 8) * N + nblk + col : nullptr;
#pragma unroll
    for (int r = 0; r < 8; ++r) {
        size_t roff = (size_t)r * N;               // D frag: VGPR r -> M = r + 8*half
        float v0 = acc0[r], v1 = acc1[r], v2 = acc2[r], v3 = acc3[r];
        if (Rbase) {
            v0 += Rbase[roff];
            v1 += Rbase[roff + 16];
            v2 += Rbase[roff + 32];
            v3 += Rbase[roff + 48];
        }
        Cbase[roff]      = v0;
        Cbase[roff + 16] = v1;
        Cbase[roff + 32] = v2;
        Cbase[roff + 48] = v3;
    }
}

// ---------------------------------------------------------------------------
// Depthwise causal conv1d (k=4) + SiLU over xi = xz[..., :512]
// ---------------------------------------------------------------------------
__global__ void k_conv_silu(const float* __restrict__ xz, const float* __restrict__ cw,
                            const float* __restrict__ cb, float* __restrict__ xc,
                            _Float16* __restrict__ xch) {
    int i = blockIdx.x * 256 + threadIdx.x;
    if (i >= BLROWS * DINNER) return;
    int d = i % DINNER;
    int bl = i / DINNER;
    int l = bl % NTOK, b = bl / NTOK;
    float acc = cb[d];
#pragma unroll
    for (int t = 0; t < 4; ++t) {
        int ls = l - 3 + t;
        if (ls >= 0)
            acc += xz[((size_t)(b * NTOK + ls)) * (2 * DINNER) + d] * cw[d * 4 + t];
    }
    float s = acc / (1.f + expf(-acc));
    xc[i] = s;
    xch[i] = (_Float16)s;
}

// ---------------------------------------------------------------------------
// dt = softplus(dt_pre @ dt_proj_w^T + b)   (K=16, VALU)
// ---------------------------------------------------------------------------
__global__ void k_dtproj(const float* __restrict__ xdbl, const float* __restrict__ wdt,
                         const float* __restrict__ bdt, float* __restrict__ dt) {
    int i = blockIdx.x * 256 + threadIdx.x;
    if (i >= BLROWS * DINNER) return;
    int d = i % DINNER;
    int bl = i / DINNER;
    const float* row = xdbl + (size_t)bl * XDBLP;
    float acc = bdt[d];
#pragma unroll
    for (int r = 0; r < DTRANK; ++r) acc += row[r] * wdt[d * DTRANK + r];
    dt[i] = (acc > 20.f) ? acc : log1pf(expf(acc));
}

// ---------------------------------------------------------------------------
// Selective scan. 32 blocks; thread = (d_local = t>>3, sgrp = t&7) owns 8 states.
// h = exp(dt*A)h + dt*B*xc ; y = h·C + xc*D ; out = y*silu(z)  (f16)
// Prefetch (global_prefetch_b8) a few steps ahead to hide L2 latency.
// ---------------------------------------------------------------------------
__global__ void k_scan(const float* __restrict__ dt, const float* __restrict__ xc,
                       const float* __restrict__ xdbl, const float* __restrict__ xz,
                       const float* __restrict__ A_log, const float* __restrict__ Dp,
                       _Float16* __restrict__ ymul) {
    int t = threadIdx.x;
    int b = blockIdx.x >> 4;
    int d = ((blockIdx.x & 15) << 5) + (t >> 3);
    int s0 = (t & 7) * 8;

    float a8[8], h8[8];
#pragma unroll
    for (int j = 0; j < 8; ++j) {
        a8[j] = -expf(A_log[(size_t)d * DSTATE + s0 + j]);
        h8[j] = 0.f;
    }
    float Dv = Dp[d];

    for (int l = 0; l < NTOK; ++l) {
        size_t bl = (size_t)b * NTOK + l;
        if (l + 8 < NTOK) {
            size_t blp = bl + 8;
            __builtin_prefetch(xdbl + blp * XDBLP + DTRANK + s0, 0, 0);
            __builtin_prefetch(dt + blp * DINNER + d, 0, 0);
            __builtin_prefetch(xc + blp * DINNER + d, 0, 0);
        }
        float dtv = dt[bl * DINNER + d];
        float xv  = xc[bl * DINNER + d];
        const float* Bp = xdbl + bl * XDBLP + DTRANK;
        const float* Cp = Bp + DSTATE;
        float yp = 0.f;
#pragma unroll
        for (int j = 0; j < 8; ++j) {
            float dA = expf(dtv * a8[j]);
            h8[j] = dA * h8[j] + dtv * Bp[s0 + j] * xv;
            yp += h8[j] * Cp[s0 + j];
        }
        yp += __shfl_down(yp, 4, 8);
        yp += __shfl_down(yp, 2, 8);
        yp += __shfl_down(yp, 1, 8);
        if ((t & 7) == 0) {
            float y = yp + xv * Dv;
            float zv = xz[bl * (2 * DINNER) + DINNER + d];
            float g = zv / (1.f + expf(-zv));
            ymul[bl * DINNER + d] = (_Float16)(y * g);
        }
    }
}

// ---------------------------------------------------------------------------
// 2048-point radix-2 FFT along N (one line per block, in LDS). ortho scale.
// If realOut != null: write residual + real part (final inverse pass).
// ---------------------------------------------------------------------------
__device__ __forceinline__ int brev11(int x) { return (int)(__brev((unsigned)x) >> 21); }

__global__ void k_fft_line(const float* __restrict__ inR, const float* __restrict__ inI,
                           float* __restrict__ outR, float* __restrict__ outI,
                           int inverse, const float* __restrict__ residual,
                           float* __restrict__ realOut) {
    __shared__ float sr[NTOK], si[NTOK];
    int line = blockIdx.x, t = threadIdx.x;
    int b = line >> 8;
    int c = line & 255;                        // channel = k*64+d
    size_t base = (size_t)b * NTOK * DIMC + c; // stride over n = 256

    for (int q = 0; q < 8; ++q) {
        int i = t + (q << 8);
        int j = brev11(i);
        sr[i] = inR[base + (size_t)j * DIMC];
        si[i] = inI ? inI[base + (size_t)j * DIMC] : 0.f;
    }
    __syncthreads();

    float sign = inverse ? 1.f : -1.f;
    for (int st = 1; st <= 11; ++st) {
        int half = 1 << (st - 1);
        for (int p = t; p < 1024; p += 256) {
            int grp = p >> (st - 1);
            int pos = p & (half - 1);
            int i0 = (grp << st) + pos;
            int i1 = i0 + half;
            float ang = sign * 6.283185307179586f * (float)pos / (float)(1 << st);
            float sn, cs;
            __sincosf(ang, &sn, &cs);
            float xr = sr[i1], xi = si[i1];
            float tr = xr * cs - xi * sn;
            float ti = xr * sn + xi * cs;
            float ur = sr[i0], ui = si[i0];
            sr[i0] = ur + tr; si[i0] = ui + ti;
            sr[i1] = ur - tr; si[i1] = ui - ti;
        }
        __syncthreads();
    }

    const float scale = 0.022097086912079608f;  // 1/sqrt(2048)
    if (realOut) {
        for (int q = 0; q < 8; ++q) {
            int i = t + (q << 8);
            size_t idx = base + (size_t)i * DIMC;
            realOut[idx] = residual[idx] + sr[i] * scale;
        }
    } else {
        for (int q = 0; q < 8; ++q) {
            int i = t + (q << 8);
            size_t idx = base + (size_t)i * DIMC;
            outR[idx] = sr[i] * scale;
            outI[idx] = si[i] * scale;
        }
    }
}

// ---------------------------------------------------------------------------
// 4-point DFT along the NB axis (ortho: *0.5). Pointwise over (b,n,d).
// ---------------------------------------------------------------------------
__global__ void k_fft4(const float* __restrict__ inR, const float* __restrict__ inI,
                       float* __restrict__ outR, float* __restrict__ outI, int inverse) {
    int i = blockIdx.x * 256 + threadIdx.x;   // over BL*64
    if (i >= BLROWS * 64) return;
    int d = i & 63;
    size_t base = (size_t)(i >> 6) * DIMC + d;
    float r0 = inR[base], i0 = inI[base];
    float r1 = inR[base + 64], i1 = inI[base + 64];
    float r2 = inR[base + 128], i2 = inI[base + 128];
    float r3 = inR[base + 192], i3 = inI[base + 192];
    float s = inverse ? 1.f : -1.f;
    outR[base]       = 0.5f * (r0 + r1 + r2 + r3);
    outI[base]       = 0.5f * (i0 + i1 + i2 + i3);
    outR[base + 64]  = 0.5f * (r0 - s * i1 - r2 + s * i3);
    outI[base + 64]  = 0.5f * (i0 + s * r1 - i2 - s * r3);
    outR[base + 128] = 0.5f * (r0 - r1 + r2 - r3);
    outI[base + 128] = 0.5f * (i0 - i1 + i2 - i3);
    outR[base + 192] = 0.5f * (r0 + s * i1 - r2 - s * i3);
    outI[base + 192] = 0.5f * (i0 - s * r1 - i2 + s * r3);
}

// ---------------------------------------------------------------------------
// EinFFT complex block-diagonal mixers: relu -> softshrink. Block = one (b,n).
// thread t -> (k = t>>6, o = t&63)
// ---------------------------------------------------------------------------
__global__ void k_einmix(const float* __restrict__ vr, const float* __restrict__ vi,
                         const float* __restrict__ cw1, const float* __restrict__ cb1,
                         const float* __restrict__ cw2, const float* __restrict__ cb2,
                         float* __restrict__ outR, float* __restrict__ outI) {
    int bn = blockIdx.x, t = threadIdx.x;
    int k = t >> 6, o = t & 63;
    __shared__ float xrs[DIMC], xis[DIMC], r1s[DIMC], i1s[DIMC];
    size_t rb = (size_t)bn * DIMC;
    xrs[t] = vr[rb + t];
    xis[t] = vi[rb + t];
    __syncthreads();

    float ar = cb1[k * 64 + o];
    float ai = cb1[256 + k * 64 + o];
    const float* w1r = cw1 + (size_t)k * 4096 + o;          // cw1[0][k][d][o]
    const float* w1i = cw1 + (size_t)(4 + k) * 4096 + o;    // cw1[1][k][d][o]
#pragma unroll 4
    for (int dd = 0; dd < 64; ++dd) {
        float xr = xrs[k * 64 + dd], xi = xis[k * 64 + dd];
        float wr = w1r[dd * 64], wi = w1i[dd * 64];
        ar += xr * wr - xi * wi;
        ai += xr * wi + xi * wr;
    }
    r1s[t] = fmaxf(ar, 0.f);
    i1s[t] = fmaxf(ai, 0.f);
    __syncthreads();

    float br = cb2[k * 64 + o];
    float bi = cb2[256 + k * 64 + o];
    const float* w2r = cw2 + (size_t)k * 4096 + o;
    const float* w2i = cw2 + (size_t)(4 + k) * 4096 + o;
#pragma unroll 4
    for (int dd = 0; dd < 64; ++dd) {
        float xr = r1s[k * 64 + dd], xi = i1s[k * 64 + dd];
        float wr = w2r[dd * 64], wi = w2i[dd * 64];
        br += xr * wr - xi * wi;
        bi += xr * wi + xi * wr;
    }
    br = (br > LAMBDA_S) ? br - LAMBDA_S : ((br < -LAMBDA_S) ? br + LAMBDA_S : 0.f);
    bi = (bi > LAMBDA_S) ? bi - LAMBDA_S : ((bi < -LAMBDA_S) ? bi + LAMBDA_S : 0.f);
    outR[rb + t] = br;
    outI[rb + t] = bi;
}

// ---------------------------------------------------------------------------
extern "C" void kernel_launch(void* const* d_in, const int* in_sizes, int n_in,
                              void* d_out, int out_size, void* d_ws, size_t ws_size,
                              hipStream_t stream) {
    (void)in_sizes; (void)n_in; (void)out_size; (void)ws_size;
    const float* x        = (const float*)d_in[0];
    const float* n1w      = (const float*)d_in[1];
    const float* n1b      = (const float*)d_in[2];
    const float* in_proj  = (const float*)d_in[3];   // (1024, 256)
    const float* conv_w   = (const float*)d_in[4];   // (512, 1, 4)
    const float* conv_b   = (const float*)d_in[5];
    const float* x_proj   = (const float*)d_in[6];   // (144, 512)
    const float* dt_w     = (const float*)d_in[7];   // (512, 16)
    const float* dt_b     = (const float*)d_in[8];
    const float* A_log    = (const float*)d_in[9];
    const float* Dp       = (const float*)d_in[10];
    const float* out_proj = (const float*)d_in[11];  // (256, 512)
    const float* n2w      = (const float*)d_in[12];
    const float* n2b      = (const float*)d_in[13];
    const float* cw1      = (const float*)d_in[14];
    const float* cb1      = (const float*)d_in[15];
    const float* cw2      = (const float*)d_in[16];
    const float* cb2      = (const float*)d_in[17];
    float* out            = (float*)d_out;

    // ---- workspace carve-up ----
    char* p = (char*)d_ws;
    auto take = [&](size_t bytes) -> char* {
        char* r = p;
        p += (bytes + 255) & ~(size_t)255;
        return r;
    };
    _Float16* wt_in  = (_Float16*)take((size_t)DIMC * 1024 * 2);
    _Float16* wt_xp  = (_Float16*)take((size_t)DINNER * XDBLP * 2);
    _Float16* wt_out = (_Float16*)take((size_t)DINNER * DIMC * 2);
    _Float16* xn_h   = (_Float16*)take((size_t)BLROWS * DIMC * 2);
    float*    xz     = (float*)take((size_t)BLROWS * 1024 * 4);
    float*    xc     = (float*)take((size_t)BLROWS * DINNER * 4);
    _Float16* xc_h   = (_Float16*)take((size_t)BLROWS * DINNER * 2);
    float*    xdbl   = (float*)take((size_t)BLROWS * XDBLP * 4);
    float*    dt     = (float*)take((size_t)BLROWS * DINNER * 4);
    _Float16* ymul   = (_Float16*)take((size_t)BLROWS * DINNER * 2);
    float*    x2     = (float*)take((size_t)BLROWS * DIMC * 4);
    float*    fr     = (float*)take((size_t)BLROWS * DIMC * 4);
    float*    fi     = (float*)take((size_t)BLROWS * DIMC * 4);
    float*    gr     = (float*)take((size_t)BLROWS * DIMC * 4);
    float*    gi     = (float*)take((size_t)BLROWS * DIMC * 4);

    // ---- weight prep (f16 transposed, x_proj zero-padded to 192 cols) ----
    k_transpose_f16<<<(DIMC * 1024 + 255) / 256, 256, 0, stream>>>(in_proj, wt_in,
                                                                   1024, DIMC, 1024);
    k_transpose_f16<<<(DINNER * XDBLP + 255) / 256, 256, 0, stream>>>(x_proj, wt_xp,
                                                                      XDBLC, DINNER, XDBLP);
    k_transpose_f16<<<(DINNER * DIMC + 255) / 256, 256, 0, stream>>>(out_proj, wt_out,
                                                                     DIMC, DINNER, DIMC);

    // ---- Mamba branch ----
    k_layernorm<<<BLROWS, 256, 0, stream>>>(x, n1w, n1b, xn_h, nullptr);
    k_gemm_wmma<<<dim3(BLROWS / 128, 1024 / 64), 256, 0, stream>>>(xn_h, wt_in, xz, nullptr,
                                                                   BLROWS, 1024, DIMC);
    k_conv_silu<<<(BLROWS * DINNER + 255) / 256, 256, 0, stream>>>(xz, conv_w, conv_b, xc, xc_h);
    k_gemm_wmma<<<dim3(BLROWS / 128, XDBLP / 64), 256, 0, stream>>>(xc_h, wt_xp, xdbl,
                                                                    nullptr, BLROWS,
                                                                    XDBLP, DINNER);
    k_dtproj<<<(BLROWS * DINNER + 255) / 256, 256, 0, stream>>>(xdbl, dt_w, dt_b, dt);
    k_scan<<<BATCH * (DINNER / 32), 256, 0, stream>>>(dt, xc, xdbl, xz, A_log, Dp, ymul);
    k_gemm_wmma<<<dim3(BLROWS / 128, DIMC / 64), 256, 0, stream>>>(ymul, wt_out, x2, x,
                                                                   BLROWS, DIMC, DINNER);

    // ---- EinFFT branch ----
    k_layernorm<<<BLROWS, 256, 0, stream>>>(x2, n2w, n2b, nullptr, fr);
    k_fft_line<<<BATCH * DIMC, 256, 0, stream>>>(fr, nullptr, gr, gi, 0, nullptr, nullptr);
    k_fft4<<<(BLROWS * 64 + 255) / 256, 256, 0, stream>>>(gr, gi, fr, fi, 0);
    k_einmix<<<BLROWS, 256, 0, stream>>>(fr, fi, cw1, cb1, cw2, cb2, gr, gi);
    k_fft4<<<(BLROWS * 64 + 255) / 256, 256, 0, stream>>>(gr, gi, fr, fi, 1);
    k_fft_line<<<BATCH * DIMC, 256, 0, stream>>>(fr, fi, nullptr, nullptr, 1, x2, out);
}
